// Attention_Head_82626580840557
// MI455X (gfx1250) — matrix operand order repaired and checked by reading, not compiled, not run on hardware
//
#include <hip/hip_runtime.h>
#include <hip/hip_bf16.h>

#define EMB   1024
#define SEQ   2048
#define NBAT  8
#define HEADD 64
#define RANK  8
#define ROWS  (NBAT * SEQ)          // 16384
#define WEFF_ELEMS (HEADD * EMB)    // 65536 per projection

typedef __attribute__((ext_vector_type(16))) __bf16 v16bf;
typedef __attribute__((ext_vector_type(8)))  float  v8f;

union FragB16 {
    v16bf   v;
    unsigned u[8];
    __bf16   h[16];
};

__device__ __forceinline__ v8f wmma_bf16(const FragB16 &a, const FragB16 &b, v8f c) {
    return __builtin_amdgcn_wmma_f32_16x16x32_bf16(
        /*neg_a=*/false, a.v, /*neg_b=*/false, b.v,
        /*c_mod=*/(short)0, c, /*reuse_a=*/false, /*reuse_b=*/false);
}

// ---------------------------------------------------------------------------
// Kernel 1: fold LoRA into effective weights, bf16 output.
// W_eff[p][h][e] = W[h][e] + 2 * sum_r B[h][r] * A[r][e]
// ---------------------------------------------------------------------------
__global__ void weff_kernel(const float* __restrict__ Wq, const float* __restrict__ Wk,
                            const float* __restrict__ Wv,
                            const float* __restrict__ Aq, const float* __restrict__ Bq,
                            const float* __restrict__ Ak, const float* __restrict__ Bk,
                            const float* __restrict__ Av, const float* __restrict__ Bv,
                            __bf16* __restrict__ weff) {
    int i = blockIdx.x * blockDim.x + threadIdx.x;
    if (i >= 3 * WEFF_ELEMS) return;
    int p   = i >> 16;          // projection
    int rem = i & (WEFF_ELEMS - 1);
    int h   = rem >> 10;
    int e   = rem & (EMB - 1);
    const float* W = (p == 0) ? Wq : ((p == 1) ? Wk : Wv);
    const float* A = (p == 0) ? Aq : ((p == 1) ? Ak : Av);
    const float* B = (p == 0) ? Bq : ((p == 1) ? Bk : Bv);
    float acc = W[h * EMB + e];
#pragma unroll
    for (int r = 0; r < RANK; ++r)
        acc += 2.0f * B[h * RANK + r] * A[r * EMB + e];
    weff[i] = (__bf16)acc;
}

// ---------------------------------------------------------------------------
// Kernel 2: fused QKV projection GEMM. One wave computes 16 rows of q,k,v.
// A fragment (x tile, fp32->bf16) is shared across all 12 WMMAs per K-step.
// q,k stored bf16 row-major [row,64]; v stored transposed [b][64][2048].
// ---------------------------------------------------------------------------
__global__ void qkv_kernel(const float* __restrict__ X, const __bf16* __restrict__ weff,
                           __bf16* __restrict__ qb, __bf16* __restrict__ kb,
                           __bf16* __restrict__ vT) {
    const int lane  = threadIdx.x & 31;
    const int wave  = threadIdx.x >> 5;
    const int tile  = blockIdx.x * 8 + wave;       // 0..1023
    const int row0  = tile << 4;                   // 16 rows per wave
    const int rA    = lane & 15;
    const int kbase = (lane >> 4) << 3;            // 0 or 8
    const int koff  = (lane >> 4) << 4;            // 0 or 16 (B frags)
    const int ncol  = lane & 15;
    const int half  = lane >> 4;

    v8f acc[3][4];
#pragma unroll
    for (int p = 0; p < 3; ++p)
#pragma unroll
        for (int j2 = 0; j2 < 4; ++j2) acc[p][j2] = (v8f)0.0f;

    const float* xrow = X + (size_t)(row0 + rA) * EMB;

    for (int e0 = 0; e0 < EMB; e0 += 32) {
        // A fragment: 16x32 slice of X, converted fp32 -> bf16
        FragB16 a;
#pragma unroll
        for (int j = 0; j < 8; ++j) {
            int kk = koff + (j << 1);
            float2 f = *(const float2*)(xrow + e0 + kk);
            a.h[2 * j]     = (__bf16)f.x;
            a.h[2 * j + 1] = (__bf16)f.y;
        }
#pragma unroll
        for (int p = 0; p < 3; ++p) {
#pragma unroll
            for (int j2 = 0; j2 < 4; ++j2) {
                int h = (j2 << 4) + ncol;
                const __bf16* wrow = weff + p * WEFF_ELEMS + h * EMB + e0 + koff;
                FragB16 b;
#pragma unroll
                for (int j = 0; j < 8; ++j)
                    b.u[j] = *(const unsigned*)(wrow + 2 * j);
                acc[p][j2] = wmma_bf16(a, b, acc[p][j2]);
            }
        }
    }

    // Store. C/D layout: element r, lane -> row = r + 8*half, col = 16*j2 + ncol.
#pragma unroll
    for (int p = 0; p < 3; ++p) {
#pragma unroll
        for (int j2 = 0; j2 < 4; ++j2) {
#pragma unroll
            for (int r = 0; r < 8; ++r) {
                int row = row0 + r + (half << 3);
                int col = (j2 << 4) + ncol;
                __bf16 bv = (__bf16)acc[p][j2][r];
                if (p == 0)      qb[(size_t)row * HEADD + col] = bv;
                else if (p == 1) kb[(size_t)row * HEADD + col] = bv;
                else {
                    int b = row >> 11, s = row & (SEQ - 1);
                    vT[((size_t)((b << 6) + col) << 11) + s] = bv;
                }
            }
        }
    }
}

// ---------------------------------------------------------------------------
// Kernel 3: flash attention with async Global->LDS K/V staging.
// A block owns 8 query tiles (128 queries) of ONE batch; K/V blocks are
// staged cooperatively into a double-buffered LDS tile with
// global_load_async_to_lds_b128 (ASYNCcnt), overlapping the WMMA compute
// of the current block with the copy of the next one.
// ---------------------------------------------------------------------------
__global__ void attn_kernel(const __bf16* __restrict__ qb, const __bf16* __restrict__ kb,
                            const __bf16* __restrict__ vT, const int* __restrict__ mrk,
                            float* __restrict__ out) {
    // [buf][0..2047]  = K tile, 32 keys x 64 head, row-major (4 KB)
    // [buf][2048..]   = V tile, 64 head x 32 keys (Vt layout)   (4 KB)
    __shared__ __align__(16) __bf16 kvbuf[2][4096];
    __shared__ __align__(16) __bf16 plds[8 * 16 * 32];   // per-wave 16x32 P tile

    const int tid   = threadIdx.x;
    const int lane  = tid & 31;
    const int wave  = tid >> 5;
    const int b     = blockIdx.x >> 4;                   // 16 blocks per batch
    const int q0    = (((blockIdx.x & 15) << 3) + wave) << 4;
    const int rA    = lane & 15;
    const int kbase = (lane >> 4) << 3;
    const int koff  = (lane >> 4) << 4;
    const int ncol  = lane & 15;
    const int half  = lane >> 4;
    const float NEG_INF = -__builtin_inff();

    // ---- async-copy bases (per-thread, 16 B each per tile per block) ----
    const __bf16* kglob = kb + (((size_t)b << 11) * HEADD);   // batch K base
    const __bf16* vglob = vT + ((size_t)b << 17);             // batch Vt base (64*2048)
    const unsigned k_voff0 = (unsigned)tid * 16u;             // + n0*128 bytes
    const int      vh      = tid >> 2, vq = tid & 3;          // 4 threads per Vt row
    const unsigned v_voff0 = (unsigned)(vh * 4096 + vq * 16); // + n0*2 bytes
    unsigned lds_k[2], lds_v[2];
#pragma unroll
    for (int i = 0; i < 2; ++i) {
        lds_k[i] = (unsigned)(uintptr_t)(&kvbuf[i][0])    + (unsigned)tid * 16u;
        lds_v[i] = (unsigned)(uintptr_t)(&kvbuf[i][2048]) + (unsigned)(vh * 64 + vq * 16);
    }

#define ISSUE_KV(N0, BUF)                                                          \
    do {                                                                           \
        asm volatile("global_load_async_to_lds_b128 %0, %1, %2"                    \
                     :: "v"(lds_k[BUF]), "v"(k_voff0 + (unsigned)(N0) * 128u),     \
                        "s"(kglob) : "memory");                                    \
        asm volatile("global_load_async_to_lds_b128 %0, %1, %2"                    \
                     :: "v"(lds_v[BUF]), "v"(v_voff0 + (unsigned)(N0) * 2u),       \
                        "s"(vglob) : "memory");                                    \
    } while (0)

    // ---- Q fragments (head dim split into two K=32 chunks) ----
    FragB16 aq[2];
    const __bf16* qrow = qb + ((size_t)((b << 11) + q0 + rA)) * HEADD;
#pragma unroll
    for (int f = 0; f < 2; ++f)
#pragma unroll
        for (int j = 0; j < 8; ++j) {
            int kk = (f << 5) + koff + (j << 1);
            aq[f].u[j] = *(const unsigned*)(qrow + kk);
        }

    v8f o[4];
#pragma unroll
    for (int j2 = 0; j2 < 4; ++j2) o[j2] = (v8f)0.0f;
    float rmax[8], rsum[8];
#pragma unroll
    for (int r = 0; r < 8; ++r) { rmax[r] = NEG_INF; rsum[r] = 0.0f; }

    const int* mb  = mrk + (b << 11);
    __bf16*    myp = plds + wave * (16 * 32);

    ISSUE_KV(0, 0);   // prologue: stage first K/V block

    for (int n0 = 0; n0 < SEQ; n0 += 32) {
        const int cur = (n0 >> 5) & 1;

        // Block n0 resident in LDS (own asynccnt) + all waves done with the
        // buffer the next copy will overwrite.
        asm volatile("s_wait_asynccnt 0x0" ::: "memory");
        __syncthreads();
        if (n0 + 32 < SEQ) ISSUE_KV(n0 + 32, cur ^ 1);

        const __bf16* Kt = &kvbuf[cur][0];
        const __bf16* Vt = &kvbuf[cur][2048];

        // ---- scores: S = Q * K^T, two 16x16 tiles (key halves), from LDS ----
        v8f s0 = (v8f)0.0f, s1 = (v8f)0.0f;
#pragma unroll
        for (int kh = 0; kh < 2; ++kh) {
            const __bf16* krow = Kt + ((kh << 4) + ncol) * HEADD + koff;
            FragB16 b0, b1;
#pragma unroll
            for (int j = 0; j < 8; ++j) {
                b0.u[j] = *(const unsigned*)(krow + 2 * j);
                b1.u[j] = *(const unsigned*)(krow + 32 + 2 * j);
            }
            if (kh == 0) { s0 = wmma_bf16(aq[0], b0, s0); s0 = wmma_bf16(aq[1], b1, s0); }
            else         { s1 = wmma_bf16(aq[0], b0, s1); s1 = wmma_bf16(aq[1], b1, s1); }
        }

        const int m0 = mb[n0 + ncol];
        const int m1 = mb[n0 + 16 + ncol];

        // ---- online softmax update (rows live in half-waves of 16 lanes) ----
#pragma unroll
        for (int r = 0; r < 8; ++r) {
            float x0 = m0 ? s0[r] * 0.125f : NEG_INF;
            float x1 = m1 ? s1[r] * 0.125f : NEG_INF;
            float nm = fmaxf(rmax[r], fmaxf(x0, x1));
            nm = fmaxf(nm, __shfl_xor(nm, 1, 32));
            nm = fmaxf(nm, __shfl_xor(nm, 2, 32));
            nm = fmaxf(nm, __shfl_xor(nm, 4, 32));
            nm = fmaxf(nm, __shfl_xor(nm, 8, 32));
            bool dead  = (nm == NEG_INF);
            float corr = dead ? 1.0f : __expf(rmax[r] - nm);
            float p0   = dead ? 0.0f : __expf(x0 - nm);
            float p1   = dead ? 0.0f : __expf(x1 - nm);
            rmax[r] = nm;
            float ps = p0 + p1;
            ps += __shfl_xor(ps, 1, 32);
            ps += __shfl_xor(ps, 2, 32);
            ps += __shfl_xor(ps, 4, 32);
            ps += __shfl_xor(ps, 8, 32);
            rsum[r] = rsum[r] * corr + ps;
#pragma unroll
            for (int j2 = 0; j2 < 4; ++j2) o[j2][r] *= corr;
            int m = r + (half << 3);
            myp[m * 32 + ncol]      = (__bf16)p0;
            myp[m * 32 + 16 + ncol] = (__bf16)p1;
        }

        // wave-private LDS P tile: enforce store->load ordering
        asm volatile("s_wait_dscnt 0" ::: "memory");

        // ---- O += P * V (V fed from LDS Vt tile) ----
        FragB16 ap;
#pragma unroll
        for (int j = 0; j < 8; ++j) {
            int kk = koff + (j << 1);
            ap.u[j] = *(const unsigned*)(myp + rA * 32 + kk);
        }
#pragma unroll
        for (int j2 = 0; j2 < 4; ++j2) {
            const __bf16* vrow = Vt + ((j2 << 4) + ncol) * 32 + koff;
            FragB16 bv;
#pragma unroll
            for (int j = 0; j < 8; ++j)
                bv.u[j] = *(const unsigned*)(vrow + 2 * j);
            o[j2] = wmma_bf16(ap, bv, o[j2]);
        }
    }
#undef ISSUE_KV

    // ---- epilogue: divide by running sum (nan_to_num -> 0 for dead rows) ----
#pragma unroll
    for (int j2 = 0; j2 < 4; ++j2)
#pragma unroll
        for (int r = 0; r < 8; ++r) {
            float denom = rsum[r];
            float val = (denom > 0.0f) ? o[j2][r] / denom : 0.0f;
            int row = (b << 11) + q0 + r + (half << 3);
            int col = (j2 << 4) + ncol;
            out[(size_t)row * HEADD + col] = val;
        }
}

// ---------------------------------------------------------------------------
extern "C" void kernel_launch(void* const* d_in, const int* in_sizes, int n_in,
                              void* d_out, int out_size, void* d_ws, size_t ws_size,
                              hipStream_t stream) {
    const float* X   = (const float*)d_in[0];
    const int*   mrk = (const int*)d_in[1];
    const float* Wq  = (const float*)d_in[2];
    const float* Wk  = (const float*)d_in[3];
    const float* Wv  = (const float*)d_in[4];
    const float* Aq  = (const float*)d_in[5];
    const float* Bq  = (const float*)d_in[6];
    const float* Ak  = (const float*)d_in[7];
    const float* Bk  = (const float*)d_in[8];
    const float* Av  = (const float*)d_in[9];
    const float* Bv  = (const float*)d_in[10];
    float* out = (float*)d_out;

    // Workspace layout (bf16):
    //   weff: 3 * 64 * 1024 | q: 16384*64 | k: 16384*64 | vT: 8*64*2048
    __bf16* weff = (__bf16*)d_ws;
    __bf16* qb   = weff + 3 * WEFF_ELEMS;
    __bf16* kb   = qb + (size_t)ROWS * HEADD;
    __bf16* vT   = kb + (size_t)ROWS * HEADD;

    weff_kernel<<<(3 * WEFF_ELEMS + 255) / 256, 256, 0, stream>>>(
        Wq, Wk, Wv, Aq, Bq, Ak, Bk, Av, Bv, weff);

    // 1024 row-tiles of 16, 8 waves per block
    qkv_kernel<<<128, 256, 0, stream>>>(X, weff, qb, kb, vT);

    // 1024 query-tiles of 16, 8 waves per block (one batch per block)
    attn_kernel<<<128, 256, 0, stream>>>(qb, kb, vT, mrk, out);
}
